// WGAT_14508399525899
// MI455X (gfx1250) — matrix operand order repaired and checked by reading, not compiled, not run on hardware
//
#include <hip/hip_runtime.h>
#include <hip/hip_bf16.h>
#include <math.h>

typedef _Float16 half_t;
typedef __attribute__((ext_vector_type(16))) _Float16 v16h;
typedef __attribute__((ext_vector_type(8)))  float    v8f;
typedef __attribute__((ext_vector_type(4)))  unsigned int u32x4;
typedef __attribute__((ext_vector_type(8)))  int      i32x8;
typedef __attribute__((ext_vector_type(4)))  int      i32x4;

#define C_    128
#define H_    192
#define W_    192
#define WIN_  8
#define N1_   24          // windows per side
#define NWIN_ 576         // N1_*N1_
#define SP_   64          // WIN_*WIN_
#define KNN_  8
#define DF_   8192        // C_*SP_  (elements per window)
#define WPSZ  147456      // 9*8*4*32*16 packed f16 weight elements

// ---------------------------------------------------------------------------
// TDM: DMA one 16KB f16 window (8192 contiguous elements) from global to LDS.
// D# built per CDNA5 ISA 8.3/8.4: group0 = {count=1, lds_addr, global_addr,
// type=2}; group1 = {data_size=2B, tensor_dim0=tile_dim0=8192, dims1=1,
// stride0=8192}. Issued from one wave only (EXEC is ignored by TDM, so the
// guard must be a scalar branch); completion via TENSORcnt.
// ---------------------------------------------------------------------------
__device__ __forceinline__ void tdm_load_window(uint64_t gaddr, uint32_t laddr) {
    u32x4 g0;
    g0[0] = 1u;                                           // count=1, user desc
    g0[1] = (uint32_t)__builtin_amdgcn_readfirstlane((int)laddr);
    g0[2] = (uint32_t)__builtin_amdgcn_readfirstlane((int)(uint32_t)gaddr);
    g0[3] = ((uint32_t)__builtin_amdgcn_readfirstlane((int)(uint32_t)(gaddr >> 32)))
            | (2u << 30);                                 // type=2 ("image")
    i32x8 g1;
    g1[0] = (1 << 16);                                    // data_size = 2 bytes
    g1[1] = (int)((unsigned)DF_ << 16);                   // tensor_dim0 = 8192
    g1[2] = (1 << 16);                                    // tensor_dim1 = 1
    g1[3] = (int)((unsigned)DF_ << 16);                   // tile_dim0 = 8192
    g1[4] = 1;                                            // tile_dim1 = 1
    g1[5] = DF_;                                          // tensor_dim0_stride
    g1[6] = 0;
    g1[7] = 0;
    i32x4 z4 = {0, 0, 0, 0};
#if __clang_major__ >= 23
    i32x8 z8 = {0, 0, 0, 0, 0, 0, 0, 0};
    __builtin_amdgcn_tensor_load_to_lds(g0, g1, z4, z4, z8, 0);
#else
    __builtin_amdgcn_tensor_load_to_lds(g0, g1, z4, z4, 0);
#endif
}

// ---------------------------------------------------------------------------
// 1) per-window squared norms (fp32)
// ---------------------------------------------------------------------------
__global__ void wgat_sqnorm(const float* __restrict__ x, float* __restrict__ sq) {
    int bw = blockIdx.x;                 // b*NWIN_ + n
    int b  = bw / NWIN_, n = bw % NWIN_;
    int wy0 = (n / N1_) * WIN_, wx0 = (n % N1_) * WIN_;
    float acc = 0.f;
    for (int d = threadIdx.x; d < DF_; d += blockDim.x) {
        int ci = d >> 6, s = d & 63;
        int y = wy0 + (s >> 3), xq = wx0 + (s & 7);
        float v = x[((size_t)(b * C_ + ci) * H_ + y) * W_ + xq];
        acc = fmaf(v, v, acc);
    }
    __shared__ float red[256];
    red[threadIdx.x] = acc; __syncthreads();
    for (int off = 128; off > 0; off >>= 1) {
        if (threadIdx.x < off) red[threadIdx.x] += red[threadIdx.x + off];
        __syncthreads();
    }
    if (threadIdx.x == 0) sq[bw] = red[0];
}

// ---------------------------------------------------------------------------
// 2) pairwise euclidean distance, 16x16 tile per block, fp32 (exactness for knn)
// ---------------------------------------------------------------------------
__global__ void wgat_dist(const float* __restrict__ x, const float* __restrict__ sq,
                          float* __restrict__ D) {
    int b  = blockIdx.z;
    int i0 = blockIdx.y * 16, j0 = blockIdx.x * 16;
    int tx = threadIdx.x & 15, ty = threadIdx.x >> 4;
    __shared__ float Ai[16][64];
    __shared__ float Aj[16][64];
    float acc = 0.f;
    for (int ci = 0; ci < C_; ++ci) {
        #pragma unroll
        for (int r = 0; r < 4; ++r) {
            int tt  = threadIdx.x * 4 + r;      // 0..1023
            int row = tt >> 6, col = tt & 63;
            int ni = i0 + row, nj = j0 + row;
            int yi = (ni / N1_) * WIN_ + (col >> 3), xi = (ni % N1_) * WIN_ + (col & 7);
            int yj = (nj / N1_) * WIN_ + (col >> 3), xj = (nj % N1_) * WIN_ + (col & 7);
            Ai[row][col] = x[((size_t)(b * C_ + ci) * H_ + yi) * W_ + xi];
            Aj[row][col] = x[((size_t)(b * C_ + ci) * H_ + yj) * W_ + xj];
        }
        __syncthreads();
        #pragma unroll 8
        for (int d = 0; d < 64; ++d) acc = fmaf(Ai[ty][d], Aj[tx][d], acc);
        __syncthreads();
    }
    float d2 = sq[b * NWIN_ + i0 + ty] + sq[b * NWIN_ + j0 + tx] - 2.0f * acc;
    D[((size_t)b * NWIN_ + (i0 + ty)) * NWIN_ + (j0 + tx)] = sqrtf(fmaxf(d2, 0.f));
}

// ---------------------------------------------------------------------------
// 3) knn (k=8 smallest, ascending) + edge scores
// ---------------------------------------------------------------------------
__global__ void wgat_knn(const float* __restrict__ D, int* __restrict__ idx,
                         float* __restrict__ score, int total) {
    int t = blockIdx.x * blockDim.x + threadIdx.x;
    if (t >= total) return;
    const float* row = D + (size_t)t * NWIN_;
    int   chosen[KNN_];
    float dist[KNN_];
    for (int k = 0; k < KNN_; ++k) {
        float best = 3.4e38f; int bi = 0;
        for (int j = 0; j < NWIN_; ++j) {
            float v = row[j];
            bool used = false;
            for (int q = 0; q < k; ++q) used |= (chosen[q] == j);
            if (!used && v < best) { best = v; bi = j; }
        }
        chosen[k] = bi; dist[k] = best;
    }
    float sigma = 0.f;
    for (int k = 0; k < KNN_; ++k) sigma += dist[k];
    sigma *= (1.0f / KNN_);
    float inv_s2 = 1.0f / (sigma * sigma);
    for (int k = 0; k < KNN_; ++k) {
        idx[t * KNN_ + k]   = chosen[k];
        score[t * KNN_ + k] = __expf(-dist[k] * inv_s2);
    }
}

// ---------------------------------------------------------------------------
// 4) pack OIHW fp32 weights into the 16-bit A-matrix (16x32) lane layout:
//    Wp[tap][coTile(8)][kt(4)][lane(32)][h(16)]  (f16)
// ---------------------------------------------------------------------------
__global__ void wgat_packw(const float* __restrict__ Wsrc, half_t* __restrict__ Wp) {
    int t = blockIdx.x * blockDim.x + threadIdx.x;
    if (t >= WPSZ) return;
    int h    = t & 15;
    int lane = (t >> 4)  & 31;
    int kt   = (t >> 9)  & 3;
    int coT  = (t >> 11) & 7;
    int tap  = t >> 14;
    int hi = lane >> 4;
    int kk = hi * 16 + h;
    int ci = kt * 32 + kk;
    int co = coT * 16 + (lane & 15);
    int ky = tap / 3, kx = tap % 3;
    Wp[t] = (half_t)Wsrc[(((size_t)co * C_ + ci) * 3 + ky) * 3 + kx];
}

// ---------------------------------------------------------------------------
// 5) pack NCHW fp32 x into windowed-transposed f16 [b][n][spatial][ci]
//    (the exact byte image the TDM copies and the B-fragments consume)
// ---------------------------------------------------------------------------
__global__ void wgat_pack16(const float* __restrict__ x, half_t* __restrict__ x16,
                            int total) {
    int t = blockIdx.x * blockDim.x + threadIdx.x;
    if (t >= total) return;
    int sci = t & (DF_ - 1);
    int bw  = t >> 13;                  // / DF_
    int b = bw / NWIN_, n = bw % NWIN_;
    int s = sci >> 7, ci = sci & 127;
    int y  = (n / N1_) * WIN_ + (s >> 3);
    int xx = (n % N1_) * WIN_ + (s & 7);
    x16[t] = (half_t)x[((size_t)(b * C_ + ci) * H_ + y) * W_ + xx];
}

// ---------------------------------------------------------------------------
// 6) graph-conv layer. Input: f16 windowed-transposed [b][n][sp][ci].
//    Per destination window (1 block, 8 waves): double-buffered TDM fetch of
//    the K=8 source windows into LDS, conv3x3 as per-tap implicit GEMM on
//    v_wmma_f32_16x16x32_f16, then score*acc + bias (conv is linear, so the
//    edge score commutes with the conv), exact GELU, running max over edges.
//    out_mode 0: f16 windowed-transposed [b][n][sp][co] (feeds next layer)
//    out_mode 1: f32 NCHW (final output)
// ---------------------------------------------------------------------------
__global__ __launch_bounds__(256, 2)
void wgat_conv(const half_t* __restrict__ in16,
               const int* __restrict__ idx, const float* __restrict__ score,
               const half_t* __restrict__ Wp, const float* __restrict__ bias,
               void* __restrict__ outp, int out_mode) {
    int bw   = blockIdx.x;
    int b    = bw / NWIN_, n = bw % NWIN_;
    int wave = threadIdx.x >> 5;       // 0..7 -> co tile
    int lane = threadIdx.x & 31;
    int hi   = lane >> 4;

    __shared__ __align__(128) half_t win[2][SP_][C_];   // double-buffered tile
    __shared__ __align__(128) half_t zrow[C_];          // zero halo row
    for (int q = threadIdx.x; q < C_; q += 256) zrow[q] = (half_t)0.f;

    // scalar (wave-uniform) guard: only wave 0 drives the TDM
    bool wv0 = (__builtin_amdgcn_readfirstlane((int)threadIdx.x) < 32);

    if (wv0) {
        int src0 = __builtin_amdgcn_readfirstlane(idx[bw * KNN_]);
        tdm_load_window((uint64_t)(uintptr_t)(in16 + ((size_t)b * NWIN_ + src0) * DF_),
                        (uint32_t)(uintptr_t)&win[0][0][0]);
    }

    v8f best[4];
    #pragma unroll
    for (int nt = 0; nt < 4; ++nt)
        #pragma unroll
        for (int r = 0; r < 8; ++r) best[nt][r] = -3.4e38f;

    for (int e = 0; e < KNN_; ++e) {
        if (wv0) {
            if (e + 1 < KNN_) {   // prefetch next edge into the other buffer
                int srcn = __builtin_amdgcn_readfirstlane(idx[bw * KNN_ + e + 1]);
                tdm_load_window(
                    (uint64_t)(uintptr_t)(in16 + ((size_t)b * NWIN_ + srcn) * DF_),
                    (uint32_t)(uintptr_t)&win[(e + 1) & 1][0][0]);
                __builtin_amdgcn_s_wait_tensorcnt(1);   // current buffer ready
            } else {
                __builtin_amdgcn_s_wait_tensorcnt(0);
            }
        }
        __syncthreads();                                // publish LDS to all waves

        const half_t (*buf)[C_] = win[e & 1];
        v8f acc[4] = {};
        for (int tap = 0; tap < 9; ++tap) {
            int dy = tap / 3 - 1, dx = tap % 3 - 1;
            v16h afrag[4];
            #pragma unroll
            for (int kt = 0; kt < 4; ++kt) {
                const v16h* ap = (const v16h*)(Wp +
                    ((((size_t)tap * 8 + wave) * 4 + kt) * 32 + lane) * 16);
                afrag[kt] = *ap;
            }
            #pragma unroll
            for (int nt = 0; nt < 4; ++nt) {
                int s  = nt * 16 + (lane & 15);
                int sy = (s >> 3) + dy, sx = (s & 7) + dx;
                bool valid = (sy >= 0) & (sy < WIN_) & (sx >= 0) & (sx < WIN_);
                const half_t* rp = valid ? &buf[sy * WIN_ + sx][0] : &zrow[0];
                #pragma unroll
                for (int kt = 0; kt < 4; ++kt) {
                    v16h bfrag = *(const v16h*)(rp + kt * 32 + hi * 16);
                    acc[nt] = __builtin_amdgcn_wmma_f32_16x16x32_f16(
                        false, afrag[kt], false, bfrag, (short)0, acc[nt],
                        false, false);
                }
            }
        }
        // score * acc + bias, exact GELU, max over edges
        float sc = score[bw * KNN_ + e];
        #pragma unroll
        for (int nt = 0; nt < 4; ++nt) {
            #pragma unroll
            for (int r = 0; r < 8; ++r) {
                int   co = wave * 16 + r + 8 * hi;
                float v  = fmaf(sc, acc[nt][r], bias[co]);
                v = 0.5f * v * (1.0f + erff(v * 0.70710678118f));
                best[nt][r] = fmaxf(best[nt][r], v);
            }
        }
        __syncthreads();    // everyone done reading buf before TDM reuses it
    }

    #pragma unroll
    for (int nt = 0; nt < 4; ++nt) {
        #pragma unroll
        for (int r = 0; r < 8; ++r) {
            int co = wave * 16 + r + 8 * hi;
            int s  = nt * 16 + (lane & 15);
            float v = best[nt][r];
            if (out_mode == 0) {
                ((half_t*)outp)[(((size_t)bw) * SP_ + s) * C_ + co] = (half_t)v;
            } else {
                int y  = (n / N1_) * WIN_ + (s >> 3);
                int xx = (n % N1_) * WIN_ + (s & 7);
                ((float*)outp)[((size_t)(b * C_ + co) * H_ + y) * W_ + xx] = v;
            }
        }
    }
}

// ---------------------------------------------------------------------------
extern "C" void kernel_launch(void* const* d_in, const int* in_sizes, int n_in,
                              void* d_out, int out_size, void* d_ws, size_t ws_size,
                              hipStream_t stream) {
    const float* x  = (const float*)d_in[0];
    const float* W1 = (const float*)d_in[1];
    const float* b1 = (const float*)d_in[2];
    const float* W2 = (const float*)d_in[3];
    const float* b2 = (const float*)d_in[4];
    float* out = (float*)d_out;
    int nb = in_sizes[0] / (C_ * H_ * W_);   // batch (=2)

    char* p = (char*)d_ws;
    auto alloc = [&](size_t bytes) -> char* {
        char* r = p;
        p += (bytes + 255) & ~(size_t)255;
        return r;
    };
    float*  sq    = (float*) alloc((size_t)nb * NWIN_ * 4);
    float*  D     = (float*) alloc((size_t)nb * NWIN_ * NWIN_ * 4);
    int*    idx   = (int*)   alloc((size_t)nb * NWIN_ * KNN_ * 4);
    float*  score = (float*) alloc((size_t)nb * NWIN_ * KNN_ * 4);
    half_t* Wp1   = (half_t*)alloc((size_t)WPSZ * 2);
    half_t* Wp2   = (half_t*)alloc((size_t)WPSZ * 2);
    half_t* x16   = (half_t*)alloc((size_t)nb * NWIN_ * DF_ * 2);
    half_t* h116  = (half_t*)alloc((size_t)nb * NWIN_ * DF_ * 2);

    wgat_sqnorm<<<nb * NWIN_, 256, 0, stream>>>(x, sq);
    wgat_dist<<<dim3(NWIN_ / 16, NWIN_ / 16, nb), 256, 0, stream>>>(x, sq, D);
    wgat_knn<<<(nb * NWIN_ + 127) / 128, 128, 0, stream>>>(D, idx, score, nb * NWIN_);
    wgat_packw<<<(WPSZ + 255) / 256, 256, 0, stream>>>(W1, Wp1);
    wgat_packw<<<(WPSZ + 255) / 256, 256, 0, stream>>>(W2, Wp2);
    int tot16 = nb * NWIN_ * DF_;
    wgat_pack16<<<(tot16 + 255) / 256, 256, 0, stream>>>(x, x16, tot16);

    // layer 1: TDM-gather from x16, write f16 windowed-transposed h116
    wgat_conv<<<nb * NWIN_, 256, 0, stream>>>(x16, idx, score, Wp1, b1, h116, 0);
    // layer 2: TDM-gather from h116, write f32 NCHW output directly
    wgat_conv<<<nb * NWIN_, 256, 0, stream>>>(h116, idx, score, Wp2, b2, out, 1);
}
